// SuperGATConv_36679020708591
// MI455X (gfx1250) — matrix operand order repaired and checked by reading, not compiled, not run on hardware
//
#include <hip/hip_runtime.h>
#include <hip/hip_bf16.h>
#include <math.h>

#define IN_DIM 128
#define KH 8        // heads
#define DH 16       // dim per head
#define OUTF 128    // KH*DH
#define NEG_SLOPE 0.2f
#define RT 4        // row tiles per wave (64 rows per block)

typedef __attribute__((ext_vector_type(16))) __bf16 v16bf;
typedef __attribute__((ext_vector_type(8)))  float  v8f;

// ---------------------------------------------------------------------------
// Kernel 0: one-time pack of W (128x128 f32) into bf16 in exact WMMA-B lane
// order: packedB[((it*8 + head)*32 + lane)*16 + j] = bf16(W[k, col]) with
// k = it*32 + (lane>>4)*8 + (j&7) + (j>>3)*16, col = head*16 + (lane&15).
// 16384 elements (32 KB) — permanently L2/L0 resident thereafter.
// ---------------------------------------------------------------------------
__global__ void gat_packB_kernel(const float* __restrict__ W,
                                 __bf16* __restrict__ packedB) {
  const int idx = blockIdx.x * blockDim.x + threadIdx.x;
  if (idx >= 4 * KH * 32 * 16) return;
  const int j    = idx & 15;
  const int lane = (idx >> 4) & 31;
  const int h    = (idx >> 9) & 7;
  const int it   = idx >> 12;
  const int k    = it * 32 + (lane >> 4) * 8 + (j & 7) + (j >> 3) * 16;
  const int col  = h * DH + (lane & 15);
  packedB[idx] = (__bf16)W[(size_t)k * OUTF + col];
}

// ---------------------------------------------------------------------------
// Kernel 1: feat_src = feat @ W via v_wmma_f32_16x16x32_bf16 with fused
// per-head attention dot products (el, er). Block = 256 threads = 8 waves;
// wave w owns head w's 16-column slice; each wave processes RT=4 row tiles
// (block covers 64 rows), reusing the single B operand per k-step across all
// 4 tiles -> 16 WMMAs per wave, B loads amortized.
// ---------------------------------------------------------------------------
__global__ __launch_bounds__(256)
void gat_gemm_attn_kernel(const float* __restrict__ feat,
                          const __bf16* __restrict__ packedB,
                          const float* __restrict__ attn_l,
                          const float* __restrict__ attn_r,
                          float* __restrict__ feat_src,
                          float* __restrict__ el,
                          float* __restrict__ er,
                          int Nn) {
  const int lane = threadIdx.x & 31;
  const int wave = threadIdx.x >> 5;   // head index / column tile
  const int half = lane >> 4;          // 0: lanes 0-15, 1: lanes 16-31
  const int l16  = lane & 15;
  const int row0 = blockIdx.x * (16 * RT);
  const int col  = wave * 16 + l16;    // B / D column

  v8f acc[RT];
#pragma unroll
  for (int t = 0; t < RT; ++t) acc[t] = (v8f){};

#pragma unroll
  for (int it = 0; it < 4; ++it) {
    // B operand: one contiguous 32-byte vector load per lane (L0/L2-hot).
    const v16bf b =
        *(const v16bf*)(packedB + (((size_t)(it * KH + wave) * 32 + lane) << 4));

    // A tiles: 16x32 bf16 each (lanes0-15: K=[0..7]+[16..23],
    // lanes16-31: K=[8..15]+[24..31], packed pairs per VGPR).
    const int ka = it * 32 + half * 8;
#pragma unroll
    for (int t = 0; t < RT; ++t) {
      const int row_raw = row0 + t * 16 + l16;
      const int row_a   = row_raw < Nn ? row_raw : Nn - 1;  // clamp loads
      const float* ap = feat + (size_t)row_a * IN_DIM + ka;
      v16bf a;
#pragma unroll
      for (int j = 0; j < 8; ++j) {
        a[j]     = (__bf16)ap[j];
        a[8 + j] = (__bf16)ap[16 + j];
      }
      acc[t] = __builtin_amdgcn_wmma_f32_16x16x32_bf16(
          /*neg_a=*/false, a, /*neg_b=*/false, b,
          /*c_mod=*/(short)0, acc[t], /*reuse_a=*/false, /*reuse_b=*/false);
    }
  }

  // Epilogue: store D tiles + reduce el/er per row (head 'wave').
  // acc[t][r] holds row (t*16 + r + 8*half), column l16 of the tile.
  const float wl = attn_l[wave * DH + l16];
  const float wr = attn_r[wave * DH + l16];

#pragma unroll
  for (int t = 0; t < RT; ++t) {
#pragma unroll
    for (int r = 0; r < 8; ++r) {
      const int row = row0 + t * 16 + r + half * 8;
      if (row < Nn) feat_src[(size_t)row * OUTF + col] = acc[t][r];
      float tl = acc[t][r] * wl;
      float tr = acc[t][r] * wr;
#pragma unroll
      for (int m = 1; m < 16; m <<= 1) {  // reduce within each 16-lane half
        tl += __shfl_xor(tl, m, 32);
        tr += __shfl_xor(tr, m, 32);
      }
      if (l16 == 0 && row < Nn) {
        el[(size_t)row * KH + wave] = tl;
        er[(size_t)row * KH + wave] = tr;
      }
    }
  }
}

// ---------------------------------------------------------------------------
// Kernel 2: initialize segment-max to -inf, segment-sum and output to zero.
// ---------------------------------------------------------------------------
__global__ void gat_init_kernel(float* __restrict__ mbuf, float* __restrict__ sbuf,
                                float* __restrict__ out, int nk, int nkd) {
  const int i = blockIdx.x * blockDim.x + threadIdx.x;
  if (i < nk) { mbuf[i] = -INFINITY; sbuf[i] = 0.0f; }
  if (i < nkd) out[i] = 0.0f;
}

// ---------------------------------------------------------------------------
// Kernel 3: e = leaky_relu(el[src] + er[dst]); stash per-edge e and
// atomic-max into m[dst]. One thread per edge, float4-vectorized over heads.
// ---------------------------------------------------------------------------
__global__ __launch_bounds__(256)
void gat_edge_max_kernel(const int* __restrict__ src, const int* __restrict__ dst,
                         const float* __restrict__ el, const float* __restrict__ er,
                         float* __restrict__ ebuf, float* __restrict__ mbuf, int E) {
  const int e = blockIdx.x * blockDim.x + threadIdx.x;
  if (e >= E) return;
  const int s = src[e], d = dst[e];
  const float4* elp = (const float4*)(el + (size_t)s * KH);
  const float4* erp = (const float4*)(er + (size_t)d * KH);
  const float4 l0 = elp[0], l1 = elp[1];
  const float4 r0 = erp[0], r1 = erp[1];
  float v[KH] = { l0.x + r0.x, l0.y + r0.y, l0.z + r0.z, l0.w + r0.w,
                  l1.x + r1.x, l1.y + r1.y, l1.z + r1.z, l1.w + r1.w };
  float* mb = mbuf + (size_t)d * KH;
#pragma unroll
  for (int k = 0; k < KH; ++k) {
    v[k] = v[k] > 0.0f ? v[k] : NEG_SLOPE * v[k];
    atomicMax(mb + k, v[k]);            // global_atomic_max_num_f32, L2-resident
  }
  float4* ep = (float4*)(ebuf + (size_t)e * KH);
  ep[0] = make_float4(v[0], v[1], v[2], v[3]);
  ep[1] = make_float4(v[4], v[5], v[6], v[7]);
}

// ---------------------------------------------------------------------------
// Kernel 4: ex = exp(e - m[dst]); stash ex; atomic-add into s[dst].
// ---------------------------------------------------------------------------
__global__ __launch_bounds__(256)
void gat_edge_exp_kernel(const int* __restrict__ dst,
                         float* __restrict__ ebuf,
                         const float* __restrict__ mbuf,
                         float* __restrict__ sbuf, int E) {
  const int e = blockIdx.x * blockDim.x + threadIdx.x;
  if (e >= E) return;
  const int d = dst[e];
  float4* ep = (float4*)(ebuf + (size_t)e * KH);
  const float4* mp = (const float4*)(mbuf + (size_t)d * KH);
  const float4 e0 = ep[0], e1 = ep[1];
  const float4 m0 = mp[0], m1 = mp[1];
  float ex[KH] = { __expf(e0.x - m0.x), __expf(e0.y - m0.y),
                   __expf(e0.z - m0.z), __expf(e0.w - m0.w),
                   __expf(e1.x - m1.x), __expf(e1.y - m1.y),
                   __expf(e1.z - m1.z), __expf(e1.w - m1.w) };
  float* sb = sbuf + (size_t)d * KH;
#pragma unroll
  for (int k = 0; k < KH; ++k) atomicAdd(sb + k, ex[k]);
  ep[0] = make_float4(ex[0], ex[1], ex[2], ex[3]);
  ep[1] = make_float4(ex[4], ex[5], ex[6], ex[7]);
}

// ---------------------------------------------------------------------------
// Kernel 5: out[dst] += feat_src[src] * (ex / s[dst]). 32 lanes per edge;
// lane owns 4 contiguous floats, all inside head k = lane/4 so it needs a
// single attention coefficient. Gather + atomics stay in the 192MB L2.
// ---------------------------------------------------------------------------
__global__ __launch_bounds__(256)
void gat_edge_agg_kernel(const int* __restrict__ src, const int* __restrict__ dst,
                         const float* __restrict__ feat_src,
                         const float* __restrict__ ebuf,
                         const float* __restrict__ sbuf,
                         float* __restrict__ out, int E) {
  const int e = blockIdx.x * 8 + (threadIdx.x >> 5);
  const int lane = threadIdx.x & 31;
  if (e >= E) return;
  const int s = src[e], d = dst[e];
  const int k = lane >> 2;   // head for this lane's 4 elements
  const float a = ebuf[(size_t)e * KH + k] / sbuf[(size_t)d * KH + k];
  const float4 f = ((const float4*)(feat_src + (size_t)s * OUTF))[lane];
  float* op = out + (size_t)d * OUTF + lane * 4;
  atomicAdd(op + 0, f.x * a);
  atomicAdd(op + 1, f.y * a);
  atomicAdd(op + 2, f.z * a);
  atomicAdd(op + 3, f.w * a);
}

// ---------------------------------------------------------------------------
extern "C" void kernel_launch(void* const* d_in, const int* in_sizes, int n_in,
                              void* d_out, int out_size, void* d_ws, size_t ws_size,
                              hipStream_t stream) {
  const float* feat   = (const float*)d_in[0];
  const float* W      = (const float*)d_in[1];
  const float* attn_l = (const float*)d_in[2];
  const float* attn_r = (const float*)d_in[3];
  const int*   src    = (const int*)d_in[4];
  const int*   dst    = (const int*)d_in[5];
  float* out = (float*)d_out;

  const int Nn = in_sizes[0] / IN_DIM;   // 100000
  const int E  = in_sizes[4];            // 1600000

  // Workspace layout: packed bf16 B first (32 KB), then float buffers
  // (~115 MB total).
  __bf16* packedB = (__bf16*)d_ws;
  float* ws = (float*)((char*)d_ws + 4 * KH * 32 * 16 * sizeof(__bf16));
  float* feat_src = ws;  ws += (size_t)Nn * OUTF;   // N*128
  float* el       = ws;  ws += (size_t)Nn * KH;     // N*8
  float* er       = ws;  ws += (size_t)Nn * KH;
  float* mbuf     = ws;  ws += (size_t)Nn * KH;
  float* sbuf     = ws;  ws += (size_t)Nn * KH;
  float* ebuf     = ws;  ws += (size_t)E * KH;      // E*8

  const int nk  = Nn * KH;
  const int nkd = Nn * OUTF;
  const int nB  = 4 * KH * 32 * 16;                 // 16384 packed B elements

  gat_packB_kernel<<<(nB + 255) / 256, 256, 0, stream>>>(W, packedB);
  gat_gemm_attn_kernel<<<(Nn + 16 * RT - 1) / (16 * RT), 256, 0, stream>>>(
      feat, packedB, attn_l, attn_r, feat_src, el, er, Nn);
  gat_init_kernel<<<(nkd + 255) / 256, 256, 0, stream>>>(mbuf, sbuf, out, nk, nkd);
  gat_edge_max_kernel<<<(E + 255) / 256, 256, 0, stream>>>(
      src, dst, el, er, ebuf, mbuf, E);
  gat_edge_exp_kernel<<<(E + 255) / 256, 256, 0, stream>>>(
      dst, ebuf, mbuf, sbuf, E);
  gat_edge_agg_kernel<<<(E + 7) / 8, 256, 0, stream>>>(
      src, dst, feat_src, ebuf, sbuf, out, E);
}
